// RotarySelfAttention_28312424415324
// MI455X (gfx1250) — matrix operand order repaired and checked by reading, not compiled, not run on hardware
//
#include <hip/hip_runtime.h>
#include <hip/hip_bf16.h>
#include <cstdint>

// ---------------------------------------------------------------------------
// Problem constants (from reference): B=2, L=2048, D=2048, H=16, DH=128
// ---------------------------------------------------------------------------
static constexpr int B_  = 2;
static constexpr int L_  = 2048;
static constexpr int D_  = 2048;
static constexpr int H_  = 16;
static constexpr int DH_ = 128;
static constexpr int ND_ = 3 * D_;   // 6144 qkv columns

typedef __attribute__((ext_vector_type(16))) __bf16 v16bf;
typedef __attribute__((ext_vector_type(8)))  __bf16 v8bf;
typedef __attribute__((ext_vector_type(8)))  float  v8f;
typedef __attribute__((ext_vector_type(4)))  unsigned int uint4v;
typedef __attribute__((ext_vector_type(8)))  int    int8v;
typedef __attribute__((ext_vector_type(4)))  int    int4v;

__device__ __forceinline__ v16bf cat8(v8bf lo, v8bf hi) {
    return __builtin_shufflevector(lo, hi, 0,1,2,3,4,5,6,7,8,9,10,11,12,13,14,15);
}
__device__ __forceinline__ v8f v8fz() {
    v8f z = {0.f,0.f,0.f,0.f,0.f,0.f,0.f,0.f};
    return z;
}
#define WMMA_BF16(a, b, c) \
    __builtin_amdgcn_wmma_f32_16x16x32_bf16(false, (a), false, (b), (short)0, (c), false, false)

// LDS tile strides (bf16 elems). 40 => 80B row pitch: 16-byte aligned and
// 20-bank stagger so 16-row fragment reads hit distinct bank groups.
static constexpr int TS = 40;

// ---------------------------------------------------------------------------
// Tensor Data Mover: load a 2D bf16 tile (tile_d0 elems wide x tile_d1 rows)
// from global (row stride = stride_elems) into LDS at byte offset lds_off,
// inserting 16B of padding after every 64B row => 80B LDS pitch == TS*2.
// D# packing per CDNA5 ISA.
// ---------------------------------------------------------------------------
__device__ __forceinline__ void tdm_load_tile_bf16(
    unsigned lds_off, unsigned long long gaddr,
    unsigned tile_d0, unsigned tile_d1,
    unsigned tensor_d0, unsigned tensor_d1, unsigned stride_elems)
{
    uint4v g0;
    g0[0] = 1u;                                               // count=1 (valid, user mode)
    g0[1] = lds_off;                                          // lds_addr (bytes)
    g0[2] = (unsigned)(gaddr & 0xFFFFFFFFu);                  // global_addr[31:0]
    g0[3] = (unsigned)((gaddr >> 32) & 0x01FFFFFFu)           // global_addr[56:32]
          | (2u << 30);                                       // type = 2 ("image")

    int8v g1;
    g1[0] = (int)((1u << 16)      // data_size = 2 bytes
                | (1u << 20)      // pad_enable
                | (3u << 22)      // pad_interval: 16 DWORDs (64B) between pads
                | (3u << 25));    // pad_amount: 4 DWORDs (16B)
    g1[1] = (int)((tensor_d0 & 0xFFFFu) << 16);               // tensor_dim0[15:0] @ [63:48]
    g1[2] = (int)((tensor_d0 >> 16) | ((tensor_d1 & 0xFFFFu) << 16)); // dim0 hi | dim1 lo
    g1[3] = (int)((tensor_d1 >> 16) | (tile_d0 << 16));       // dim1 hi | tile_dim0
    g1[4] = (int)(tile_d1);                                   // tile_dim1 (tile_dim2 = 0)
    g1[5] = (int)(stride_elems);                              // tensor_dim0_stride[31:0]
    g1[6] = 0;                                                // stride hi | dim1_stride lo
    g1[7] = 0;

    int4v gz = {0, 0, 0, 0};
#if defined(__clang_major__) && (__clang_major__ >= 23)
    int8v z8 = {0, 0, 0, 0, 0, 0, 0, 0};
    __builtin_amdgcn_tensor_load_to_lds(g0, g1, gz, gz, z8, 0);
#else
    __builtin_amdgcn_tensor_load_to_lds(g0, g1, gz, gz, 0);
#endif
}

// ===========================================================================
// Kernel 1: qkv = x @ Wqkv + b, fused RoPE + RMSNorm epilogue.
//   Tile 128(M tokens) x 128(N cols); N tile == one head section of q/k/v.
//   Outputs: qbuf/kbuf bf16 [B,H,L,DH] (post rope+rms), vT bf16 [B,H,DH,L].
// ===========================================================================
__global__ __launch_bounds__(256) void qkv_rope_kernel(
    const float* __restrict__ x, const float* __restrict__ Wqkv,
    const float* __restrict__ bqkv, const float* __restrict__ qnw,
    const float* __restrict__ knw,
    __bf16* __restrict__ qbuf, __bf16* __restrict__ kbuf, __bf16* __restrict__ vT)
{
    extern __shared__ __align__(16) unsigned char smem[];
    __bf16* sA = (__bf16*)smem;                    // [128][TS]  x tile (row, k)
    __bf16* sB = (__bf16*)(smem + 128 * TS * 2);   // [128][TS]  W tile (n, k) transposed
    float*  sE = (float*)smem;                     // [128][132] fp32 epilogue tile (overlaps)

    const int tid  = threadIdx.x;
    const int lane = tid & 31;
    const int wv   = tid >> 5;
    const int n0   = blockIdx.x * 128;
    const int m0   = blockIdx.y * 128;
    const int wM   = (wv >> 1) * 32;   // 4 waves along M
    const int wN   = (wv & 1) * 64;    // 2 waves along N

    v8f acc[2][4];
#pragma unroll
    for (int mf = 0; mf < 2; ++mf)
#pragma unroll
        for (int nf = 0; nf < 4; ++nf) acc[mf][nf] = v8fz();

    for (int k0 = 0; k0 < D_; k0 += 32) {
        __syncthreads();
        // stage A: 128x32 fp32 -> bf16 LDS
#pragma unroll
        for (int it = 0; it < 4; ++it) {
            int idx = it * 256 + tid;
            int row = idx >> 3;
            int c4  = (idx & 7) * 4;
            float4 v = *(const float4*)(x + (size_t)(m0 + row) * D_ + k0 + c4);
            __bf16* d = sA + row * TS + c4;
            d[0] = (__bf16)v.x; d[1] = (__bf16)v.y; d[2] = (__bf16)v.z; d[3] = (__bf16)v.w;
        }
        // stage B transposed: 32x128 fp32 -> bf16 LDS [n][k]
#pragma unroll
        for (int it = 0; it < 4; ++it) {
            int idx = it * 256 + tid;
            int kk  = idx >> 5;
            int c4  = (idx & 31) * 4;
            float4 v = *(const float4*)(Wqkv + (size_t)(k0 + kk) * ND_ + n0 + c4);
            sB[(c4 + 0) * TS + kk] = (__bf16)v.x;
            sB[(c4 + 1) * TS + kk] = (__bf16)v.y;
            sB[(c4 + 2) * TS + kk] = (__bf16)v.z;
            sB[(c4 + 3) * TS + kk] = (__bf16)v.w;
        }
        if (k0 + 32 < D_) {  // gfx1250 global_prefetch_b8 of next tiles
            __builtin_prefetch(x + (size_t)(m0 + (tid >> 1)) * D_ + k0 + 32, 0, 0);
            __builtin_prefetch(Wqkv + (size_t)(k0 + 32 + (tid >> 3)) * ND_ + n0, 0, 0);
        }
        __syncthreads();

        v16bf af[2], bf_[4];
#pragma unroll
        for (int mf = 0; mf < 2; ++mf) {
            const __bf16* p = sA + (wM + mf * 16 + (lane & 15)) * TS + ((lane >> 4) * 16);
            af[mf] = cat8(*(const v8bf*)p, *(const v8bf*)(p + 8));
        }
#pragma unroll
        for (int nf = 0; nf < 4; ++nf) {
            const __bf16* p = sB + (wN + nf * 16 + (lane & 15)) * TS + ((lane >> 4) * 16);
            bf_[nf] = cat8(*(const v8bf*)p, *(const v8bf*)(p + 8));
        }
#pragma unroll
        for (int mf = 0; mf < 2; ++mf)
#pragma unroll
            for (int nf = 0; nf < 4; ++nf)
                acc[mf][nf] = WMMA_BF16(af[mf], bf_[nf], acc[mf][nf]);
    }

    // spill fp32 accumulators to LDS so epilogue threads see full DH rows
    __syncthreads();
#pragma unroll
    for (int mf = 0; mf < 2; ++mf)
#pragma unroll
        for (int nf = 0; nf < 4; ++nf)
#pragma unroll
            for (int r = 0; r < 8; ++r) {
                int row = wM + mf * 16 + r + 8 * (lane >> 4);
                int col = wN + nf * 16 + (lane & 15);
                sE[row * 132 + col] = acc[mf][nf][r];
            }
    __syncthreads();

    const int b   = m0 / L_;
    const int l0  = m0 % L_;           // tile rows are one contiguous l-range
    const int sec = n0 >> 11;          // 0=q, 1=k, 2=v
    const int h   = (n0 & 2047) >> 7;  // head within section

    if (sec == 2) {
        // v: bias add, convert, store transposed [B,H,DH,L]
        int d  = tid >> 1;
        int hf = tid & 1;
        float bias = bqkv[n0 + d];
        __bf16* dst = vT + ((size_t)((b * H_ + h) * DH_ + d)) * L_ + l0 + hf * 64;
#pragma unroll
        for (int j8 = 0; j8 < 8; ++j8) {
            v8bf t;
#pragma unroll
            for (int j = 0; j < 8; ++j)
                t[j] = (__bf16)(sE[(hf * 64 + j8 * 8 + j) * 132 + d] + bias);
            *(v8bf*)(dst + j8 * 8) = t;
        }
    } else if (tid < 128) {
        // q/k: RoPE + RMSNorm over full DH=128 row, one thread per token row
        const int   row = tid;
        const int   pos = l0 + row;
        float*      er  = sE + row * 132;
        const float K64 = 13.287712379549449f / 64.f;  // log2(10000)/64
        float sumsq = 0.f;
#pragma unroll 4
        for (int i = 0; i < 64; ++i) {
            float invf = exp2f(-(float)i * K64);       // 10000^(-i/64)
            float ang  = (float)pos * invf;
            float s, c;
            __sincosf(ang, &s, &c);
            float x1 = er[i]      + bqkv[n0 + i];
            float x2 = er[i + 64] + bqkv[n0 + i + 64];
            float r1 = x1 * c - x2 * s;                // i < 64  : x*cos - rot*sin
            float r2 = x2 * c + x1 * s;                // i >= 64 : x*cos + rot*sin
            er[i] = r1; er[i + 64] = r2;
            sumsq += r1 * r1 + r2 * r2;
        }
        float rs = rsqrtf(sumsq * (1.f / 128.f) + 1e-6f);
        const float* nw  = (sec == 0) ? qnw : knw;
        __bf16*      dst = ((sec == 0) ? qbuf : kbuf) +
                           ((size_t)((b * H_ + h) * L_ + pos)) * DH_;
#pragma unroll
        for (int c8 = 0; c8 < 16; ++c8) {
            v8bf t;
#pragma unroll
            for (int j = 0; j < 8; ++j) {
                int i = c8 * 8 + j;
                t[j] = (__bf16)(er[i] * rs * nw[i]);
            }
            *(v8bf*)(dst + c8 * 8) = t;
        }
    }
}

// ===========================================================================
// Kernel 2: flash attention. One wave owns 16 query rows; 8 waves/block cover
// 128 queries of one (b,h). Streams keys in blocks of 32.
//   scores: 8x wmma (Q A-frag x K^T B-frag), online softmax via shfl_xor,
//   P transposed D->A layout through per-wave LDS, PV: 8x wmma vs vT.
// ===========================================================================
__global__ __launch_bounds__(256) void attn_kernel(
    const __bf16* __restrict__ qbuf, const __bf16* __restrict__ kbuf,
    const __bf16* __restrict__ vT, const unsigned char* __restrict__ mask,
    __bf16* __restrict__ aout)
{
    __shared__ __align__(16) __bf16 p_lds[8][16][TS];

    const int lane = threadIdx.x & 31;
    const int wv   = threadIdx.x >> 5;
    const int qt   = blockIdx.x & 15;        // L/128 = 16 query tiles
    const int bh   = blockIdx.x >> 4;
    const int b    = bh >> 4;
    const int h    = bh & 15;
    const int q0   = qt * 128 + wv * 16;

    const __bf16* qbh = qbuf + (size_t)bh * L_ * DH_;
    const __bf16* kbh = kbuf + (size_t)bh * L_ * DH_;
    const __bf16* vbh = vT   + (size_t)bh * DH_ * L_;
    const unsigned char* mk = mask + b * L_;

    // preload Q A-fragments (16 rows x 128 dh = 4 frags of K=32)
    v16bf qa[4];
#pragma unroll
    for (int s = 0; s < 4; ++s) {
        const __bf16* p = qbh + (size_t)(q0 + (lane & 15)) * DH_ + s * 32 + ((lane >> 4) * 16);
        qa[s] = cat8(*(const v8bf*)p, *(const v8bf*)(p + 8));
    }

    v8f oacc[8];
#pragma unroll
    for (int n = 0; n < 8; ++n) oacc[n] = v8fz();
    float m_[8], l_[8];
#pragma unroll
    for (int r = 0; r < 8; ++r) { m_[r] = -1e30f; l_[r] = 0.f; }

    const float scale = 0.08838834764831845f;  // 1/sqrt(128)

    for (int k0 = 0; k0 < L_; k0 += 32) {
        if (k0 + 32 < L_)
            __builtin_prefetch(kbh + (size_t)(k0 + 32 + lane) * DH_, 0, 0);

        // ---- scores: S[16q x 32k] as two 16x16 wmma accumulators ----
        v8f s0 = v8fz(), s1 = v8fz();
#pragma unroll
        for (int s = 0; s < 4; ++s) {
            const __bf16* p0 = kbh + (size_t)(k0 + (lane & 15)) * DH_ + s * 32 + ((lane >> 4) * 16);
            const __bf16* p1 = p0 + (size_t)16 * DH_;
            v16bf b0 = cat8(*(const v8bf*)p0, *(const v8bf*)(p0 + 8));
            v16bf b1 = cat8(*(const v8bf*)p1, *(const v8bf*)(p1 + 8));
            s0 = WMMA_BF16(qa[s], b0, s0);
            s1 = WMMA_BF16(qa[s], b1, s1);
        }
        // additive mask exactly as reference: bool -> float (0.0 / 1.0)
        const float mf0 = mk[k0 + (lane & 15)]      ? 1.0f : 0.0f;
        const float mf1 = mk[k0 + 16 + (lane & 15)] ? 1.0f : 0.0f;

        float rm[8];
#pragma unroll
        for (int r = 0; r < 8; ++r) {
            s0[r] = s0[r] * scale + mf0;
            s1[r] = s1[r] * scale + mf1;
            rm[r] = fmaxf(s0[r], s1[r]);
        }
        // row max across the 16-lane column group
#pragma unroll
        for (int off = 1; off < 16; off <<= 1)
#pragma unroll
            for (int r = 0; r < 8; ++r)
                rm[r] = fmaxf(rm[r], __shfl_xor(rm[r], off, 32));

        float psum[8];
#pragma unroll
        for (int r = 0; r < 8; ++r) {
            float mn   = fmaxf(m_[r], rm[r]);
            float corr = __expf(m_[r] - mn);
            s0[r] = __expf(s0[r] - mn);
            s1[r] = __expf(s1[r] - mn);
            psum[r] = s0[r] + s1[r];
            m_[r]   = mn;
            l_[r]  *= corr;
#pragma unroll
            for (int n = 0; n < 8; ++n) oacc[n][r] *= corr;
        }
#pragma unroll
        for (int off = 1; off < 16; off <<= 1)
#pragma unroll
            for (int r = 0; r < 8; ++r)
                psum[r] += __shfl_xor(psum[r], off, 32);
#pragma unroll
        for (int r = 0; r < 8; ++r) l_[r] += psum[r];

        // ---- stage P (bf16) to LDS in D-layout, read back in A-layout ----
        {
            int prow = (lane >> 4) * 8;
            int pcol = lane & 15;
#pragma unroll
            for (int r = 0; r < 8; ++r) {
                p_lds[wv][prow + r][pcol]      = (__bf16)s0[r];
                p_lds[wv][prow + r][16 + pcol] = (__bf16)s1[r];
            }
        }
        __builtin_amdgcn_wave_barrier();   // per-wave LDS is in-order; fence scheduler
        v16bf pa;
        {
            const __bf16* p = &p_lds[wv][lane & 15][(lane >> 4) * 16];
            pa = cat8(*(const v8bf*)p, *(const v8bf*)(p + 8));
        }

        // ---- PV: out[16q x 128dh] += P x V  (vT is [dh][l], contiguous keys) ----
#pragma unroll
        for (int n = 0; n < 8; ++n) {
            const __bf16* p = vbh + (size_t)(n * 16 + (lane & 15)) * L_ + k0 + ((lane >> 4) * 16);
            v16bf vb = cat8(*(const v8bf*)p, *(const v8bf*)(p + 8));
            oacc[n] = WMMA_BF16(pa, vb, oacc[n]);
        }
    }

    // finalize: divide by l, zero masked query rows, store [B,L,H*DH]
    const int rbase = q0 + (lane >> 4) * 8;
#pragma unroll
    for (int r = 0; r < 8; ++r) {
        int   row = rbase + r;
        float inv = 1.f / l_[r];
        if (mk[row]) inv = 0.f;            // padding_mask zeroing from reference
        __bf16* orow = aout + ((size_t)(b * L_ + row)) * D_ + h * DH_;
#pragma unroll
        for (int n = 0; n < 8; ++n)
            orow[n * 16 + (lane & 15)] = (__bf16)(oacc[n][r] * inv);
    }
}

// ===========================================================================
// Kernel 3: out = attn_out(bf16) @ Wout + bout   -> fp32 d_out
//   A-tile (raw bf16) is fetched by the Tensor Data Mover into padded LDS
//   (pad 16B per 64B row -> 80B pitch == TS). One wave issues the TDM; it
//   overlaps the other waves' fp32->bf16 staging of the W tile.
// ===========================================================================
__global__ __launch_bounds__(256) void outproj_kernel(
    const __bf16* __restrict__ ain, const float* __restrict__ Wout,
    const float* __restrict__ bout, float* __restrict__ out)
{
    __shared__ __align__(16) __bf16 sA[128 * TS];
    __shared__ __align__(16) __bf16 sB[128 * TS];

    const int tid  = threadIdx.x;
    const int lane = tid & 31;
    const int wv   = tid >> 5;
    const int n0   = blockIdx.x * 128;
    const int m0   = blockIdx.y * 128;
    const int wM   = (wv >> 1) * 32;
    const int wN   = (wv & 1) * 64;

    const unsigned sA_base = (unsigned)(uintptr_t)(void*)sA;

    v8f acc[2][4];
#pragma unroll
    for (int mf = 0; mf < 2; ++mf)
#pragma unroll
        for (int nf = 0; nf < 4; ++nf) acc[mf][nf] = v8fz();

    for (int k0 = 0; k0 < D_; k0 += 32) {
        __syncthreads();   // previous iteration's fragment reads complete
        // ---- A tile via Tensor Data Mover (wave 0 issues; TENSORcnt) ----
        if (wv == 0) {
            unsigned long long ga =
                (unsigned long long)(uintptr_t)(ain + (size_t)m0 * D_ + k0);
            tdm_load_tile_bf16(sA_base, ga, /*tile_d0=*/32, /*tile_d1=*/128,
                               /*tensor_d0=*/(unsigned)(D_ - k0), /*tensor_d1=*/128,
                               /*stride_elems=*/(unsigned)D_);
        }
        // ---- B tile staged by all waves (fp32 -> bf16, transposed [n][k]) ----
#pragma unroll
        for (int it = 0; it < 4; ++it) {
            int idx = it * 256 + tid;
            int kk  = idx >> 5;
            int c4  = (idx & 31) * 4;
            float4 v = *(const float4*)(Wout + (size_t)(k0 + kk) * D_ + n0 + c4);
            sB[(c4 + 0) * TS + kk] = (__bf16)v.x;
            sB[(c4 + 1) * TS + kk] = (__bf16)v.y;
            sB[(c4 + 2) * TS + kk] = (__bf16)v.z;
            sB[(c4 + 3) * TS + kk] = (__bf16)v.w;
        }
        if (k0 + 32 < D_)
            __builtin_prefetch(Wout + (size_t)(k0 + 32 + (tid >> 3)) * D_ + n0, 0, 0);
        if (wv == 0)
            __builtin_amdgcn_s_wait_tensorcnt(0);
        __syncthreads();   // publish TDM-written sA + staged sB to all waves

        v16bf af[2], bf_[4];
#pragma unroll
        for (int mf = 0; mf < 2; ++mf) {
            const __bf16* p = sA + (wM + mf * 16 + (lane & 15)) * TS + ((lane >> 4) * 16);
            af[mf] = cat8(*(const v8bf*)p, *(const v8bf*)(p + 8));
        }
#pragma unroll
        for (int nf = 0; nf < 4; ++nf) {
            const __bf16* p = sB + (wN + nf * 16 + (lane & 15)) * TS + ((lane >> 4) * 16);
            bf_[nf] = cat8(*(const v8bf*)p, *(const v8bf*)(p + 8));
        }
#pragma unroll
        for (int mf = 0; mf < 2; ++mf)
#pragma unroll
            for (int nf = 0; nf < 4; ++nf)
                acc[mf][nf] = WMMA_BF16(af[mf], bf_[nf], acc[mf][nf]);
    }

    // epilogue: bias + fp32 store straight from D-layout
#pragma unroll
    for (int mf = 0; mf < 2; ++mf)
#pragma unroll
        for (int nf = 0; nf < 4; ++nf)
#pragma unroll
            for (int r = 0; r < 8; ++r) {
                int row = wM + mf * 16 + r + 8 * (lane >> 4);
                int col = wN + nf * 16 + (lane & 15);
                out[(size_t)(m0 + row) * D_ + n0 + col] = acc[mf][nf][r] + bout[n0 + col];
            }
}

// ===========================================================================
extern "C" void kernel_launch(void* const* d_in, const int* in_sizes, int n_in,
                              void* d_out, int out_size, void* d_ws, size_t ws_size,
                              hipStream_t stream)
{
    (void)in_sizes; (void)n_in; (void)out_size; (void)ws_size;

    const float*         x    = (const float*)d_in[0];
    const unsigned char* mask = (const unsigned char*)d_in[1];  // jnp.bool_ [B,1,1,L]
    const float*         Wqkv = (const float*)d_in[2];
    const float*         bqkv = (const float*)d_in[3];
    const float*         Wout = (const float*)d_in[4];
    const float*         bout = (const float*)d_in[5];
    const float*         qnw  = (const float*)d_in[6];
    const float*         knw  = (const float*)d_in[7];
    float*               out  = (float*)d_out;

    // workspace: 4 bf16 buffers of B*H*L*DH = 8,388,608 elems (16 MiB each)
    const size_t QBYTES = (size_t)B_ * H_ * L_ * DH_ * sizeof(__bf16);
    char* ws = (char*)d_ws;
    __bf16* qbuf = (__bf16*)(ws);
    __bf16* kbuf = (__bf16*)(ws + QBYTES);
    __bf16* vT   = (__bf16*)(ws + 2 * QBYTES);
    __bf16* aout = (__bf16*)(ws + 3 * QBYTES);

    // 1) qkv GEMM + bias + RoPE + RMSNorm, bf16 q/k/vT out
    qkv_rope_kernel<<<dim3(ND_ / 128, (B_ * L_) / 128), 256, 128 * 132 * sizeof(float), stream>>>(
        x, Wqkv, bqkv, qnw, knw, qbuf, kbuf, vT);

    // 2) flash attention: B*H*(L/128) workgroups
    attn_kernel<<<dim3(B_ * H_ * (L_ / 128)), 256, 0, stream>>>(
        qbuf, kbuf, vT, mask, aout);

    // 3) output projection -> fp32 d_out
    outproj_kernel<<<dim3(D_ / 128, (B_ * L_) / 128), 256, 0, stream>>>(
        aout, Wout, bout, out);
}
